// MTLAGL_58265526337818
// MI455X (gfx1250) — matrix operand order repaired and checked by reading, not compiled, not run on hardware
//
#include <hip/hip_runtime.h>
#include <math.h>

#define NN 50000
#define EE 800000
#define HD 128
#define NCLS 40

typedef __attribute__((ext_vector_type(16))) __bf16 v16bf;
typedef __attribute__((ext_vector_type(8)))  float  v8f;

union BFV { v16bf v; unsigned short u[16]; };

__device__ __forceinline__ unsigned short bf16rne(float f) {
  unsigned u = __float_as_uint(f);
  u += 0x7fffu + ((u >> 16) & 1u);          // round-to-nearest-even
  return (unsigned short)(u >> 16);
}

__device__ __forceinline__ float eluf(float x) { return x > 0.f ? x : (expf(x) - 1.f); }

// ---------------------------------------------------------------------------
// WMMA GEMM: Y[M x OD] = X[M x 128] @ W[OD x 128]^T (+bias) (+=Y)
// block = 256 threads = 8 wave32s; block tile = 64 rows; wave w owns cols [16w,16w+16)
// K = 128 as 4 x k32 steps of v_wmma_f32_16x16x32_bf16. OD is compile-time so the
// epilogue lowers to immediate-offset global stores.
// ---------------------------------------------------------------------------
#define AS_STRIDE 132   // ushorts; 264 B/row keeps 8B alignment for b64 LDS ops

template <int OD, bool ACC, bool BIAS>
__global__ __launch_bounds__(256) void k_gemm_nt(
    const float* __restrict__ X, const float* __restrict__ W,
    const float* __restrict__ bias, float* __restrict__ Y, int M)
{
  __shared__ unsigned short As[64 * AS_STRIDE];
  const int tid = threadIdx.x;
  const int r0  = blockIdx.x * 64;

  if (r0 + 64 < M)  // hint next row-block into cache
    __builtin_prefetch(X + (size_t)(r0 + 64) * HD + tid * 32, 0, 1);

  // stage A tile (f32 -> bf16) into LDS: float4 loads, ushort4 stores
  for (int q = tid; q < 64 * 32; q += 256) {
    int r = q >> 5, c4 = (q & 31) << 2;
    float4 xv = {0.f, 0.f, 0.f, 0.f};
    if (r0 + r < M) xv = *(const float4*)(X + (size_t)(r0 + r) * HD + c4);
    ushort4 hv;
    hv.x = bf16rne(xv.x); hv.y = bf16rne(xv.y);
    hv.z = bf16rne(xv.z); hv.w = bf16rne(xv.w);
    *(ushort4*)&As[r * AS_STRIDE + c4] = hv;
  }
  __syncthreads();

  const int wv   = tid >> 5;      // wave id == column tile
  const int lane = tid & 31;
  const int g    = lane >> 4;     // lane group (0/1)
  const int l    = lane & 15;
  const int col0 = wv * 16;
  if (col0 >= OD) return;         // wave-uniform, after the only barrier

  const int  ocol  = col0 + l;
  const bool colOK = (OD % 16 == 0) ? true : (ocol < OD);
  const int  ocl   = colOK ? ocol : (OD - 1);
  const float* wrow = W + (size_t)ocl * HD;

  // B fragments: 32x16 bf16; K = ks*32 + 16*g + j (j=0..15 consecutive), col N = l
  BFV bf[4];
#pragma unroll
  for (int ks = 0; ks < 4; ++ks) {
#pragma unroll
    for (int j4 = 0; j4 < 4; ++j4) {
      float4 wvv = {0.f, 0.f, 0.f, 0.f};
      if (colOK) wvv = *(const float4*)(wrow + ks * 32 + 16 * g + 4 * j4);
      bf[ks].u[4 * j4 + 0] = bf16rne(wvv.x);
      bf[ks].u[4 * j4 + 1] = bf16rne(wvv.y);
      bf[ks].u[4 * j4 + 2] = bf16rne(wvv.z);
      bf[ks].u[4 * j4 + 3] = bf16rne(wvv.w);
    }
  }

  float bv = 0.f;
  if (BIAS && colOK) bv = bias[ocol];

  const bool fullM = (r0 + 64 <= M);

#pragma unroll
  for (int mt = 0; mt < 4; ++mt) {
    v8f c;
#pragma unroll
    for (int v = 0; v < 8; ++v) c[v] = BIAS ? bv : 0.f;

    const unsigned short* ap = &As[(mt * 16 + l) * AS_STRIDE];
#pragma unroll
    for (int ks = 0; ks < 4; ++ks) {
      // A layout: u[0..15] = K ks*32+16g+{0..15}
      BFV af;
      *(ushort4*)&af.u[0]  = *(const ushort4*)&ap[ks * 32 + 16 * g];
      *(ushort4*)&af.u[4]  = *(const ushort4*)&ap[ks * 32 + 16 * g + 4];
      *(ushort4*)&af.u[8]  = *(const ushort4*)&ap[ks * 32 + 16 * g + 8];
      *(ushort4*)&af.u[12] = *(const ushort4*)&ap[ks * 32 + 16 * g + 12];
      c = __builtin_amdgcn_wmma_f32_16x16x32_bf16(
              false, af.v, false, bf[ks].v, (short)0, c, false, false);
    }

    // C/D layout: VGPR v -> row (v + 8*g), col l
    float* yp = Y + (size_t)(r0 + mt * 16 + 8 * g) * OD + ocol;
    if (fullM && colOK) {
#pragma unroll
      for (int v = 0; v < 8; ++v) {
        if (ACC) yp[v * OD] += c[v]; else yp[v * OD] = c[v];
      }
    } else {
#pragma unroll
      for (int v = 0; v < 8; ++v) {
        int row = r0 + mt * 16 + v + 8 * g;
        if (row < M && colOK) {
          if (ACC) yp[v * OD] += c[v]; else yp[v * OD] = c[v];
        }
      }
    }
  }
}

// ---------------------------------------------------------------------------
// small kernels
// ---------------------------------------------------------------------------
__global__ void k_weights(const float* na, const float* sc, const float* la, float* wgt) {
  if (threadIdx.x != 0 || blockIdx.x != 0) return;
  for (int r = 0; r < 3; ++r) {
    float m = -1e30f;
    for (int c = 0; c < 4; ++c) m = fmaxf(m, na[r * 4 + c]);
    float e[4], s = 0.f;
    for (int c = 0; c < 4; ++c) { e[c] = expf(na[r * 4 + c] - m); s += e[c]; }
    for (int c = 0; c < 4; ++c) wgt[r * 4 + c] = e[c] / s;
  }
  for (int r = 0; r < 2; ++r) {
    float m = fmaxf(sc[r * 2], sc[r * 2 + 1]);
    float e0 = expf(sc[r * 2] - m), e1 = expf(sc[r * 2 + 1] - m);
    wgt[12 + r * 2] = e0 / (e0 + e1); wgt[12 + r * 2 + 1] = e1 / (e0 + e1);
  }
  {
    float m = fmaxf(la[0], fmaxf(la[1], la[2]));
    float e0 = expf(la[0] - m), e1 = expf(la[1] - m), e2 = expf(la[2] - m);
    float s = e0 + e1 + e2;
    wgt[16] = e0 / s; wgt[17] = e1 / s; wgt[18] = e2 / s;
  }
}

__global__ __launch_bounds__(256) void k_degree(const int* __restrict__ dst, float* deg) {
  int e = blockIdx.x * 256 + threadIdx.x;
  if (e < EE) atomicAdd(&deg[dst[e]], 1.f);
}

__global__ __launch_bounds__(256) void k_dis(const float* deg, float* dis) {
  int n = blockIdx.x * 256 + threadIdx.x;
  if (n < NN) {
    float d = deg[n];
    dis[n] = (d > 0.f) ? rsqrtf(fmaxf(d, 1.f)) : 0.f;
  }
}

__global__ __launch_bounds__(256) void k_init_neg(float* mx) {
  int n = blockIdx.x * 256 + threadIdx.x;
  if (n < NN) mx[n] = -3.0e38f;
}

// fused GCN message scatter + neighbor-sum (shared by SAGE/GIN)
__global__ __launch_bounds__(256) void k_scatter1(
    const int* __restrict__ src, const int* __restrict__ dst,
    const float* __restrict__ dis, const float* __restrict__ hgcn,
    const float* __restrict__ cur, float* ogcn, float* agg)
{
  size_t idx = (size_t)blockIdx.x * 256 + threadIdx.x;
  if (idx >= (size_t)EE * HD) return;
  int e = (int)(idx >> 7), i = (int)(idx & 127);
  int s = src[e], d = dst[e];
  float nm = dis[s] * dis[d];
  atomicAdd(&ogcn[(size_t)d * HD + i], hgcn[(size_t)s * HD + i] * nm);
  atomicAdd(&agg [(size_t)d * HD + i], cur [(size_t)s * HD + i]);
}

// wave-per-node: sn = h . a_s, dn = h . a_d
__global__ __launch_bounds__(256) void k_gat_logits(
    const float* __restrict__ h, const float* __restrict__ a_s,
    const float* __restrict__ a_d, float* sn, float* dn)
{
  int node = blockIdx.x * 8 + (threadIdx.x >> 5);
  int lane = threadIdx.x & 31;
  if (node >= NN) return;
  const float* hp = h + (size_t)node * HD;
  float v = 0.f, u = 0.f;
  for (int j = lane; j < HD; j += 32) {
    float hv = hp[j];
    v += hv * a_s[j];
    u += hv * a_d[j];
  }
#pragma unroll
  for (int o = 16; o > 0; o >>= 1) {
    v += __shfl_xor(v, o, 32);
    u += __shfl_xor(u, o, 32);
  }
  if (lane == 0) { sn[node] = v; dn[node] = u; }
}

__device__ __forceinline__ void atomicMaxF(float* addr, float val) {
  int old = __float_as_int(*addr);
  while (__int_as_float(old) < val) {
    int prev = atomicCAS((int*)addr, old, __float_as_int(val));
    if (prev == old) break;
    old = prev;
  }
}

__global__ __launch_bounds__(256) void k_gat_max(
    const int* src, const int* dst, const float* sn, const float* dn, float* mx)
{
  int e = blockIdx.x * 256 + threadIdx.x;
  if (e >= EE) return;
  int s = src[e], d = dst[e];
  float lg = sn[s] + dn[d];
  lg = lg > 0.f ? lg : 0.2f * lg;
  atomicMaxF(&mx[d], lg);
}

__global__ __launch_bounds__(256) void k_gat_exp(
    const int* src, const int* dst, const float* sn, const float* dn,
    const float* mx, float* ee, float* zz)
{
  int e = blockIdx.x * 256 + threadIdx.x;
  if (e >= EE) return;
  int s = src[e], d = dst[e];
  float lg = sn[s] + dn[d];
  lg = lg > 0.f ? lg : 0.2f * lg;
  float ev = expf(lg - mx[d]);
  ee[e] = ev;
  atomicAdd(&zz[d], ev);
}

__global__ __launch_bounds__(256) void k_gat_scatter(
    const int* __restrict__ src, const int* __restrict__ dst,
    const float* __restrict__ ee, const float* __restrict__ zz,
    const float* __restrict__ hgat, float* ogat)
{
  size_t idx = (size_t)blockIdx.x * 256 + threadIdx.x;
  if (idx >= (size_t)EE * HD) return;
  int e = (int)(idx >> 7), i = (int)(idx & 127);
  int s = src[e], d = dst[e];
  float alpha = ee[e] / (zz[d] + 1e-16f);
  atomicAdd(&ogat[(size_t)d * HD + i], alpha * hgat[(size_t)s * HD + i]);
}

// agg -> agg/max(deg,1) in place; ginin = (1+eps)*cur + agg
__global__ __launch_bounds__(256) void k_prep(
    const float* __restrict__ cur, float* agg, const float* __restrict__ deg,
    const float* __restrict__ epsp, int l, float* ginin)
{
  size_t i = (size_t)blockIdx.x * 256 + threadIdx.x;
  if (i >= (size_t)NN * HD) return;
  int n = (int)(i >> 7);
  float a = agg[i];
  ginin[i] = (1.f + epsp[l]) * cur[i] + a;
  agg[i] = a / fmaxf(deg[n], 1.f);
}

__global__ __launch_bounds__(256) void k_combine(
    const float* o0, const float* o1, const float* o2, const float* o3,
    const float* __restrict__ wgt, int l, float* out)
{
  size_t i = (size_t)blockIdx.x * 256 + threadIdx.x;
  if (i >= (size_t)NN * HD) return;
  const float* w = wgt + l * 4;
  out[i] = w[0] * eluf(o0[i]) + w[1] * eluf(o1[i]) + w[2] * eluf(o2[i]) + w[3] * eluf(o3[i]);
}

__global__ __launch_bounds__(256) void k_finalmix(
    const float* x1, const float* x2, const float* x3,
    const float* __restrict__ wgt, float* x5)
{
  size_t i = (size_t)blockIdx.x * 256 + threadIdx.x;
  if (i >= (size_t)NN * HD) return;
  float sc0 = wgt[13], sc1 = wgt[15];                  // sc_w[0,1], sc_w[1,1]
  float la0 = wgt[16], la1 = wgt[17], la2 = wgt[18];
  float t0 = x3[i], t1 = sc0 * x1[i], t2 = sc1 * x2[i];
  float mx = fmaxf(t0, fmaxf(t1, t2));
  float sm = t0 + t1 + t2;
  float mean = sm * (1.f / 3.f);
  x5[i] = la0 * fmaxf(mx, 0.f) + la1 * fmaxf(mean, 0.f) + la2 * fmaxf(sm, 0.f);
}

// ---------------------------------------------------------------------------
extern "C" void kernel_launch(void* const* d_in, const int* in_sizes, int n_in,
                              void* d_out, int out_size, void* d_ws, size_t ws_size,
                              hipStream_t stream) {
  const float* x       = (const float*)d_in[0];
  const int*   ei      = (const int*)d_in[1];
  const float* na_a    = (const float*)d_in[2];
  const float* sc_a    = (const float*)d_in[3];
  const float* la_a    = (const float*)d_in[4];
  const float* lin1_W  = (const float*)d_in[5];
  const float* lin1_b  = (const float*)d_in[6];
  const float* gcn_W   = (const float*)d_in[7];
  const float* gcn_b   = (const float*)d_in[8];
  const float* sage_Ws = (const float*)d_in[9];
  const float* sage_Wn = (const float*)d_in[10];
  const float* gin_W   = (const float*)d_in[11];
  const float* gin_b   = (const float*)d_in[12];
  const float* gin_eps = (const float*)d_in[13];
  const float* gat_W   = (const float*)d_in[14];
  const float* gat_as  = (const float*)d_in[15];
  const float* gat_ad  = (const float*)d_in[16];
  const float* nc_W    = (const float*)d_in[17];
  const float* nc_b    = (const float*)d_in[18];
  float* out = (float*)d_out;

  const size_t NB = (size_t)NN * HD;
  float* f = (float*)d_ws;
  size_t off = 0;
  float* wgt = f + off; off += 32;
  float* deg = f + off; off += NN;
  float* dis = f + off; off += NN;
  float* sn  = f + off; off += NN;
  float* dn  = f + off; off += NN;
  float* mx  = f + off; off += NN;
  float* zz  = f + off; off += NN;
  float* ee  = f + off; off += EE;
  float* H0  = f + off; off += NB;
  float* X1  = f + off; off += NB;
  float* X2  = f + off; off += NB;
  float* X3  = f + off; off += NB;
  float* X5  = f + off; off += NB;
  float* HGC = f + off; off += NB;   // gcn h / later gin input
  float* HGT = f + off; off += NB;   // gat h
  float* OGC = f + off; off += NB;
  float* OGT = f + off; off += NB;
  float* OSG = f + off; off += NB;
  float* OGI = f + off; off += NB;
  float* AGG = f + off; off += NB;
  if (ws_size < off * sizeof(float)) return;

  const int* src = ei;
  const int* dst = ei + EE;

  const int GB  = (NN + 63) / 64;
  const int NBn = (NN + 255) / 256;
  const int EBn = (EE + 255) / 256;
  const int NHB = (int)((NB + 255) / 256);
  const int EHB = (int)(((size_t)EE * HD + 255) / 256);

  hipMemsetAsync(deg, 0, NN * sizeof(float), stream);
  k_weights<<<1, 32, 0, stream>>>(na_a, sc_a, la_a, wgt);
  k_degree<<<EBn, 256, 0, stream>>>(dst, deg);
  k_dis<<<NBn, 256, 0, stream>>>(deg, dis);

  // h = x @ lin1_W.T + lin1_b
  k_gemm_nt<HD, false, true><<<GB, 256, 0, stream>>>(x, lin1_W, lin1_b, H0, NN);

  const float* cur = H0;
  float* xs[3] = {X1, X2, X3};
  for (int l = 0; l < 3; ++l) {
    float* xo = xs[l];
    size_t wo = (size_t)l * HD * HD;

    k_gemm_nt<HD, false, true ><<<GB, 256, 0, stream>>>(cur, gcn_W + wo, gcn_b + l * HD, HGC, NN);
    k_gemm_nt<HD, false, false><<<GB, 256, 0, stream>>>(cur, gat_W + wo, nullptr,        HGT, NN);

    hipMemsetAsync(OGC, 0, NB * sizeof(float), stream);
    hipMemsetAsync(OGT, 0, NB * sizeof(float), stream);
    hipMemsetAsync(AGG, 0, NB * sizeof(float), stream);
    hipMemsetAsync(zz,  0, NN * sizeof(float), stream);
    k_init_neg<<<NBn, 256, 0, stream>>>(mx);

    k_scatter1<<<EHB, 256, 0, stream>>>(src, dst, dis, HGC, cur, OGC, AGG);

    k_gat_logits<<<(NN + 7) / 8, 256, 0, stream>>>(HGT, gat_as + l * HD, gat_ad + l * HD, sn, dn);
    k_gat_max<<<EBn, 256, 0, stream>>>(src, dst, sn, dn, mx);
    k_gat_exp<<<EBn, 256, 0, stream>>>(src, dst, sn, dn, mx, ee, zz);
    k_gat_scatter<<<EHB, 256, 0, stream>>>(src, dst, ee, zz, HGT, OGT);

    k_prep<<<NHB, 256, 0, stream>>>(cur, AGG, deg, gin_eps, l, HGC);  // HGC := gin input

    k_gemm_nt<HD, false, false><<<GB, 256, 0, stream>>>(cur, sage_Ws + wo, nullptr, OSG, NN);
    k_gemm_nt<HD, true,  false><<<GB, 256, 0, stream>>>(AGG, sage_Wn + wo, nullptr, OSG, NN);
    k_gemm_nt<HD, false, true ><<<GB, 256, 0, stream>>>(HGC, gin_W + wo, gin_b + l * HD, OGI, NN);

    k_combine<<<NHB, 256, 0, stream>>>(OGC, OSG, OGI, OGT, wgt, l, xo);
    cur = xo;
  }

  k_finalmix<<<NHB, 256, 0, stream>>>(X1, X2, X3, wgt, X5);
  k_gemm_nt<NCLS, false, true><<<GB, 256, 0, stream>>>(X5, nc_W, nc_b, out, NN);
}
